// TemporalGNN_65377992179781
// MI455X (gfx1250) — compile-verified, matrix operand order checked by reading, not run on hardware
//
#include <hip/hip_runtime.h>

#define BB 28
#define NN 500
#define TT 36
#define HH 256
#define EE 16000
#define BN (BB*NN)            // 14000
#define EXT (EE+NN)           // 16500 (edges + self loops)
#define ROWT64 ((BN + 63) / 64)  // 219 blocks of 64 rows

typedef __attribute__((ext_vector_type(16))) __bf16        v16bf;
typedef __attribute__((ext_vector_type(8)))  float         v8f;
typedef __attribute__((ext_vector_type(4)))  unsigned int  u32x4;

__device__ __forceinline__ unsigned short f2bf(float x) {
  unsigned int u = __float_as_uint(x);
  u += 0x7FFFu + ((u >> 16) & 1u);          // round-to-nearest-even
  return (unsigned short)(u >> 16);
}

// ---------------- graph norm prep ----------------
__global__ void tg_build_ext(const int* __restrict__ ei, const float* __restrict__ ew,
                             int* __restrict__ src, int* __restrict__ dst, float* __restrict__ w) {
  int i = blockIdx.x * blockDim.x + threadIdx.x;
  if (i < EE)       { src[i] = ei[i]; dst[i] = ei[EE + i]; w[i] = ew[i]; }
  else if (i < EXT) { int n = i - EE; src[i] = n; dst[i] = n; w[i] = 1.0f; }
}

__global__ void tg_deg(const int* __restrict__ dst, const float* __restrict__ w,
                       float* __restrict__ deg) {
  int i = blockIdx.x * blockDim.x + threadIdx.x;
  if (i < EXT) atomicAdd(&deg[dst[i]], w[i]);
}

__global__ void tg_norm(const int* __restrict__ src, const int* __restrict__ dst,
                        float* __restrict__ w, const float* __restrict__ deg) {
  int i = blockIdx.x * blockDim.x + threadIdx.x;
  if (i >= EXT) return;
  float ds = deg[src[i]], dd = deg[dst[i]];
  float is = ds > 0.f ? rsqrtf(ds) : 0.f;
  float id = dd > 0.f ? rsqrtf(dd) : 0.f;
  w[i] = is * w[i] * id;
}

// S[t, b*N+dst, f] += norm_e * x[b, src, f, t]   (GCN scatter folded over F_IN=2)
__global__ void tg_scatter_S(const int* __restrict__ src, const int* __restrict__ dst,
                             const float* __restrict__ nrm, const float* __restrict__ x,
                             float* __restrict__ S) {
  int idx = blockIdx.x * blockDim.x + threadIdx.x;
  if (idx >= EXT * BB) return;
  int e = idx % EXT, b = idx / EXT;
  int s = src[e], d = dst[e];
  float w = nrm[e];
  const float* xs = x + ((size_t)(b * NN + s) * 2) * TT;   // x[b,s,f,t]
  int row = b * NN + d;
  for (int t = 0; t < TT; ++t) {
    atomicAdd(&S[((size_t)t * BN + row) * 2 + 0], w * xs[t]);
    atomicAdd(&S[((size_t)t * BN + row) * 2 + 1], w * xs[TT + t]);
  }
}

// ---------------- weight prep ----------------
// Pack bottom half of Wl (rows 256..511, the H-multiplying part) into WMMA
// B-fragment order (bf16): gate slot g in {0:z,1:r,2:h}; per (kt,nt) tile 512
// bf16 elems, lane-major 16 elems/lane. K(e,kh) = e + 8*kh + (e>=8 ? 8 : 0).
__global__ void tg_pack_wfrag(const float* __restrict__ Wz, const float* __restrict__ Wr,
                              const float* __restrict__ Wh, unsigned short* __restrict__ out) {
  int i = blockIdx.x * blockDim.x + threadIdx.x;
  if (i >= 3 * 8 * 16 * 512) return;
  int g = i / 65536, rem = i % 65536;
  int tile = rem / 512, le = rem % 512;
  int kt = tile / 16, nt = tile % 16;
  int lanei = le / 16, e = le % 16;
  int kh = lanei >> 4, colL = lanei & 15;
  int K = e + 8 * kh + (e >= 8 ? 8 : 0);
  int row = 256 + kt * 32 + K;
  int col = nt * 16 + colL;
  const float* W = (g == 0) ? Wz : (g == 1) ? Wr : Wh;
  out[i] = f2bf(W[row * HH + col]);
}

// Fold GCN into per-gate M (2x256) and bconst (256)
__global__ void tg_fold(const float* Wc0, const float* bc0, const float* Wl0, const float* bl0,
                        const float* Wc1, const float* bc1, const float* Wl1, const float* bl1,
                        const float* Wc2, const float* bc2, const float* Wl2, const float* bl2,
                        float* __restrict__ M, float* __restrict__ bconst) {
  int g = blockIdx.x, j = threadIdx.x;
  const float* Wc = (g == 0) ? Wc0 : (g == 1) ? Wc1 : Wc2;
  const float* bc = (g == 0) ? bc0 : (g == 1) ? bc1 : bc2;
  const float* Wl = (g == 0) ? Wl0 : (g == 1) ? Wl1 : Wl2;
  const float* bl = (g == 0) ? bl0 : (g == 1) ? bl1 : bl2;
  float m0 = 0.f, m1 = 0.f, bb = bl[j];
  for (int k = 0; k < HH; ++k) {
    float w = Wl[k * HH + j];                 // top half of Wl
    m0 = fmaf(Wc[k], w, m0);
    m1 = fmaf(Wc[HH + k], w, m1);
    bb = fmaf(bc[k], w, bb);
  }
  M[g * 512 + j]       = m0;
  M[g * 512 + 256 + j] = m1;
  bconst[g * 256 + j]  = bb;
}

__global__ void tg_softmax(const float* __restrict__ a, float* __restrict__ o) {
  if (threadIdx.x == 0) {
    float m = a[0];
    for (int i = 1; i < TT; ++i) m = fmaxf(m, a[i]);
    float s = 0.f;
    for (int i = 0; i < TT; ++i) s += __expf(a[i] - m);
    float inv = 1.f / s;
    for (int i = 0; i < TT; ++i) o[i] = __expf(a[i] - m) * inv;
  }
}

// ---------------- recurrent step (WMMA core, 64-row block) ----------------
// Block = 64 rows x 256 cols, 8 waves. Wave w owns cols [32w, 32w+32) for all
// 4 row sub-tiles -> each B fragment fetch feeds 8 WMMAs (4x A reuse).
// All 4 A fragments of a k-step are hoisted (8 ds_load_b128 batch, immediate
// offsets from one per-lane base) and the next k-step's B fragments are
// prefetched from global under the current WMMAs.
union Bu { u32x4 q[2]; v16bf v; };
union Au { unsigned int u[8]; v16bf v; };

__device__ __forceinline__ void gate_gemm64(const unsigned int* __restrict__ hbase, // &tile[lane%16][4*(lane/16)]
                                            const unsigned short* __restrict__ wfrag,
                                            const float* __restrict__ Mg,
                                            const float* __restrict__ bc,
                                            const float (*sS)[2],
                                            int lane, int wave, v8f (&acc)[4][2]) {
  const int kh = lane >> 4;
  const int col0 = wave * 32 + (lane & 15), col1 = col0 + 16;
  const float mg00 = Mg[col0], mg10 = Mg[256 + col0], bc0 = bc[col0];
  const float mg01 = Mg[col1], mg11 = Mg[256 + col1], bc1 = bc[col1];
#pragma unroll
  for (int rt = 0; rt < 4; ++rt)
#pragma unroll
    for (int v = 0; v < 8; ++v) {
      int r = rt * 16 + v + 8 * kh;
      float s0 = sS[r][0], s1 = sS[r][1];
      acc[rt][0][v] = fmaf(s0, mg00, fmaf(s1, mg10, bc0));
      acc[rt][1][v] = fmaf(s0, mg01, fmaf(s1, mg11, bc1));
    }
  const int nt0 = wave * 2;
  Bu b0, b1, n0, n1;
  {
    const u32x4* p0 = (const u32x4*)(wfrag + nt0 * 512) + lane * 2;
    const u32x4* p1 = (const u32x4*)(wfrag + (nt0 + 1) * 512) + lane * 2;
    b0.q[0] = p0[0]; b0.q[1] = p0[1];
    b1.q[0] = p1[0]; b1.q[1] = p1[1];
  }
#pragma unroll
  for (int kt = 0; kt < 8; ++kt) {
    Au a[4];
#pragma unroll
    for (int rt = 0; rt < 4; ++rt)      // batch of 8 ds_load_b128, imm offsets
#pragma unroll
      for (int j = 0; j < 4; ++j) {
        a[rt].u[j]     = hbase[rt * 2048 + kt * 16 + j];
        a[rt].u[j + 4] = hbase[rt * 2048 + kt * 16 + 8 + j];
      }
    if (kt < 7) {   // prefetch next k-step's B fragments (global, L2-resident)
      const u32x4* p0 = (const u32x4*)(wfrag + ((kt + 1) * 16 + nt0) * 512) + lane * 2;
      const u32x4* p1 = (const u32x4*)(wfrag + ((kt + 1) * 16 + nt0 + 1) * 512) + lane * 2;
      n0.q[0] = p0[0]; n0.q[1] = p0[1];
      n1.q[0] = p1[0]; n1.q[1] = p1[1];
    }
#pragma unroll
    for (int rt = 0; rt < 4; ++rt) {
      acc[rt][0] = __builtin_amdgcn_wmma_f32_16x16x32_bf16(false, a[rt].v, false, b0.v, (short)0, acc[rt][0], false, false);
      acc[rt][1] = __builtin_amdgcn_wmma_f32_16x16x32_bf16(false, a[rt].v, false, b1.v, (short)0, acc[rt][1], false, false);
    }
    b0 = n0; b1 = n1;
  }
}

__global__ __launch_bounds__(256) void tg_step(
    const float* __restrict__ S_t,            // (BN,2) for this t
    float* __restrict__ H,                    // (BN,256) in/out (rows block-local)
    float* __restrict__ Hacc,                 // (BN,256) attention accumulator
    const unsigned short* __restrict__ Wfrag, // 3 gates x 8x16 tiles x 512 bf16
    const float* __restrict__ Mmat,           // 3 x 2 x 256
    const float* __restrict__ bconst,         // 3 x 256
    const float* __restrict__ att, int t) {
  __shared__ unsigned int   hA[64][128];      // H tile, bf16 pairs (A-frag layout)
  __shared__ unsigned short hr[64][256];      // (H*R) tile, bf16
  __shared__ float          hF[64][256];      // H tile, f32
  __shared__ float          sS[64][2];

  const int tid = threadIdx.x, lane = tid & 31, wave = tid >> 5;
  const int row0 = blockIdx.x * 64;

  for (int i = tid; i < 64 * 256; i += 256) {
    int r = i >> 8, c = i & 255;
    int gr = row0 + r;
    hF[r][c] = (gr < BN) ? H[(size_t)gr * HH + c] : 0.f;
  }
  if (tid < 128) {
    int r = tid >> 1, f = tid & 1;
    int gr = row0 + r;
    sS[r][f] = (gr < BN) ? S_t[(size_t)gr * 2 + f] : 0.f;
  }
  __syncthreads();
  for (int i = tid; i < 64 * 128; i += 256) {
    int r = i >> 7, cd = i & 127;
    hA[r][cd] = (unsigned)f2bf(hF[r][2 * cd]) | ((unsigned)f2bf(hF[r][2 * cd + 1]) << 16);
  }
  __syncthreads();

  const int kh = lane >> 4, ln = lane & 15;
  const int col0 = wave * 32 + ln, col1 = col0 + 16;
  const unsigned int* hAbase = &hA[ln][4 * kh];
  const unsigned int* hrbase = (const unsigned int*)&hr[0][0] + ln * 128 + 4 * kh;

  // ---- r gate ----
  {
    v8f racc[4][2];
    gate_gemm64(hAbase, Wfrag + 1 * 65536, Mmat + 1 * 512, bconst + 1 * 256, sS, lane, wave, racc);
#pragma unroll
    for (int rt = 0; rt < 4; ++rt)
#pragma unroll
      for (int v = 0; v < 8; ++v) {
        int r = rt * 16 + v + 8 * kh;
        float r0 = 1.f / (1.f + __expf(-racc[rt][0][v]));
        float r1 = 1.f / (1.f + __expf(-racc[rt][1][v]));
        hr[r][col0] = f2bf(hF[r][col0] * r0);
        hr[r][col1] = f2bf(hF[r][col1] * r1);
      }
  }
  __syncthreads();

  // ---- h gate (on H*R), keep Htil in registers ----
  v8f hacc[4][2];
  gate_gemm64(hrbase, Wfrag + 2 * 65536, Mmat + 2 * 512, bconst + 2 * 256, sS, lane, wave, hacc);
#pragma unroll
  for (int rt = 0; rt < 4; ++rt)
#pragma unroll
    for (int v = 0; v < 8; ++v) {
      hacc[rt][0][v] = tanhf(hacc[rt][0][v]);
      hacc[rt][1][v] = tanhf(hacc[rt][1][v]);
    }

  // ---- z gate ----
  v8f zacc[4][2];
  gate_gemm64(hAbase, Wfrag + 0 * 65536, Mmat + 0 * 512, bconst + 0 * 256, sS, lane, wave, zacc);

  // ---- GRU update + attention accumulation ----
  float a = att[t];
#pragma unroll
  for (int rt = 0; rt < 4; ++rt)
#pragma unroll
    for (int v = 0; v < 8; ++v) {
      int r = rt * 16 + v + 8 * kh;
      int gr = row0 + r;
      if (gr < BN) {
        float z0 = 1.f / (1.f + __expf(-zacc[rt][0][v]));
        float z1 = 1.f / (1.f + __expf(-zacc[rt][1][v]));
        float hn0 = z0 * hF[r][col0] + (1.f - z0) * hacc[rt][0][v];
        float hn1 = z1 * hF[r][col1] + (1.f - z1) * hacc[rt][1][v];
        size_t g0 = (size_t)gr * HH + col0;
        size_t g1 = (size_t)gr * HH + col1;
        H[g0] = hn0;           H[g1] = hn1;
        Hacc[g0] += a * hn0;   Hacc[g1] += a * hn1;
      }
    }
}

// ---------------- head: relu(concat) @ lin_W + lin_b ----------------
__global__ void tg_head(const float* __restrict__ Ht, const float* __restrict__ Hs,
                        const float* __restrict__ W, const float* __restrict__ b,
                        float* __restrict__ out) {
  int r = blockIdx.x;
  __shared__ float h[512];
  for (int i = threadIdx.x; i < 256; i += blockDim.x) {
    h[i]       = fmaxf(Ht[(size_t)r * HH + i], 0.f);
    h[256 + i] = fmaxf(Hs[(size_t)r * HH + i], 0.f);
  }
  __syncthreads();
  for (int t = threadIdx.x; t < TT; t += blockDim.x) {
    float acc = b[t];
    for (int k = 0; k < 512; ++k) acc = fmaf(h[k], W[k * TT + t], acc);
    out[(size_t)r * TT + t] = acc;
  }
}

// ---------------- launch ----------------
struct GateP { const float *Wc, *Wl, *bc, *bl; };

extern "C" void kernel_launch(void* const* d_in, const int* in_sizes, int n_in,
                              void* d_out, int out_size, void* d_ws, size_t ws_size,
                              hipStream_t stream) {
  auto F = [&](int i) { return (const float*)d_in[i]; };
  const float* x      = F(0);
  const int*   ei[2]  = { (const int*)d_in[1], (const int*)d_in[3] };
  const float* ew[2]  = { F(2), F(4) };
  const float* lin_W  = F(31);
  const float* lin_b  = F(32);

  const float* attp[2]; GateP gz[2], gr[2], gh[2];
  for (int br = 0; br < 2; ++br) {
    int base = 5 + br * 13;
    attp[br] = F(base);
    // sorted-key pytree: att,[h:Wc,Wl,bc,bl],[r],[z]; insertion order: att,[z:Wc,bc,Wl,bl],[r],[h]
    bool sorted = (in_sizes[base + 2] == 2 * HH * HH);
    GateP g[3];
    for (int i = 0; i < 3; ++i) {
      int o = base + 1 + 4 * i;
      if (sorted) g[i] = { F(o), F(o + 1), F(o + 2), F(o + 3) };
      else        g[i] = { F(o), F(o + 2), F(o + 1), F(o + 3) };
    }
    if (sorted) { gh[br] = g[0]; gr[br] = g[1]; gz[br] = g[2]; }
    else        { gz[br] = g[0]; gr[br] = g[1]; gh[br] = g[2]; }
  }

  size_t off = 0;
  auto carve = [&](size_t bytes) {
    void* p = (char*)d_ws + off;
    off = (off + bytes + 255) & ~(size_t)255;
    return p;
  };
  struct Br {
    int *src, *dst; float *nrm, *deg, *S, *M, *bc, *att, *H, *Hacc;
    unsigned short* Wf;
  } brd[2];
  for (int br = 0; br < 2; ++br) {
    brd[br].src  = (int*)carve(EXT * 4);
    brd[br].dst  = (int*)carve(EXT * 4);
    brd[br].nrm  = (float*)carve(EXT * 4);
    brd[br].deg  = (float*)carve(NN * 4);
    brd[br].S    = (float*)carve((size_t)TT * BN * 2 * 4);
    brd[br].Wf   = (unsigned short*)carve((size_t)3 * 65536 * 2);
    brd[br].M    = (float*)carve(3 * 512 * 4);
    brd[br].bc   = (float*)carve(3 * 256 * 4);
    brd[br].att  = (float*)carve(64 * 4);
    brd[br].H    = (float*)carve((size_t)BN * HH * 4);
    brd[br].Hacc = (float*)carve((size_t)BN * HH * 4);
  }

  for (int br = 0; br < 2; ++br) {
    Br& d = brd[br];
    hipMemsetAsync(d.deg, 0, NN * 4, stream);
    hipMemsetAsync(d.S, 0, (size_t)TT * BN * 2 * 4, stream);
    hipMemsetAsync(d.H, 0, (size_t)BN * HH * 4, stream);
    hipMemsetAsync(d.Hacc, 0, (size_t)BN * HH * 4, stream);
    tg_build_ext<<<(EXT + 255) / 256, 256, 0, stream>>>(ei[br], ew[br], d.src, d.dst, d.nrm);
    tg_deg<<<(EXT + 255) / 256, 256, 0, stream>>>(d.dst, d.nrm, d.deg);
    tg_norm<<<(EXT + 255) / 256, 256, 0, stream>>>(d.src, d.dst, d.nrm, d.deg);
    tg_scatter_S<<<(EXT * BB + 255) / 256, 256, 0, stream>>>(d.src, d.dst, d.nrm, x, d.S);
    tg_pack_wfrag<<<(3 * 65536 + 255) / 256, 256, 0, stream>>>(gz[br].Wl, gr[br].Wl, gh[br].Wl, d.Wf);
    tg_fold<<<3, 256, 0, stream>>>(gz[br].Wc, gz[br].bc, gz[br].Wl, gz[br].bl,
                                   gr[br].Wc, gr[br].bc, gr[br].Wl, gr[br].bl,
                                   gh[br].Wc, gh[br].bc, gh[br].Wl, gh[br].bl,
                                   d.M, d.bc);
    tg_softmax<<<1, 32, 0, stream>>>(attp[br], d.att);
  }

  for (int t = 0; t < TT; ++t)
    for (int br = 0; br < 2; ++br)
      tg_step<<<ROWT64, 256, 0, stream>>>(brd[br].S + (size_t)t * BN * 2, brd[br].H,
                                          brd[br].Hacc, brd[br].Wf, brd[br].M, brd[br].bc,
                                          brd[br].att, t);

  tg_head<<<BN, 64, 0, stream>>>(brd[0].Hacc, brd[1].Hacc, lin_W, lin_b, (float*)d_out);
}
